// ToricCRBM_84095459655731
// MI455X (gfx1250) — compile-verified, matrix operand order checked by reading, not run on hardware
//
#include <hip/hip_runtime.h>
#include <hip/hip_bf16.h>

typedef __attribute__((ext_vector_type(16))) __bf16 v16bf;
typedef __attribute__((ext_vector_type(8)))  float  v8f;

#define B_    8192
#define S_    1152
#define G_    576
#define F_    16
#define L_    24
#define NP_   576
#define KAUG  1792            // 1152 + 576 + 24 + 24, padded to mult of 32
#define KT    56              // KAUG/32
#define TILEB (B_/16)         // 512 b-tiles
#define GC    36              // G_/16 symmetry chunks
#define FRAG_ELEMS 512        // one 16x32 bf16 fragment = 32 lanes * 16 elems
#define A_TILE_ELEMS (KT*FRAG_ELEMS)      // 28672 bf16 per b-tile
#define LOOP_TILE_ELEMS (2*FRAG_ELEMS)    // 1024 bf16 (cv1/cv2 padded tiles)

// ---- workspace layout (bytes) ----
static constexpr size_t WS_A    = 0;                                       // 29,360,128
static constexpr size_t WS_AL   = WS_A  + (size_t)TILEB*A_TILE_ELEMS*2;    // +1,048,576
static constexpr size_t WS_B    = WS_AL + (size_t)TILEB*LOOP_TILE_ELEMS*2; // +33,030,144
static constexpr size_t WS_BL   = WS_B  + (size_t)G_*A_TILE_ELEMS*2;       // +1,179,648
static constexpr size_t WS_PART = WS_BL + (size_t)G_*LOOP_TILE_ELEMS*2;    // +4,718,592
static constexpr size_t WS_BIAS = WS_PART + (size_t)TILEB*GC*4*16*4;       // +32,768

// WMMA 16-bit fragment lane/position packing (ISA 7.12.2, 16-bit 16x32 layout)
__device__ __forceinline__ int frag_lane(int row, int ko) { return row + (((ko >> 3) & 1) << 4); }
__device__ __forceinline__ int frag_pos(int ko)           { return (ko & 7) + ((ko >> 4) << 3); }

// fast log(cosh(t)): |t| + log(1 + e^{-2|t|}) - log2.  The log argument is in
// (1,2], so hardware v_log_f32/v_exp_f32 are accurate enough: when e^{-2|t|}
// underflows the rounding of 1+u, the whole term is negligible vs |t|.
__device__ __forceinline__ float log_coshf(float t) {
  float a = fabsf(t);
  return a + __logf(1.0f + __expf(-2.0f * a)) - 0.69314718055994531f;
}

// ---------------- P1: build augmented activation fragments + bias --------------
__global__ void k_prep_x(const float* __restrict__ x,
                         const float* __restrict__ visible_bias,
                         const float* __restrict__ c0b, const float* __restrict__ c1b,
                         const float* __restrict__ c2b,
                         const int* __restrict__ corr0_idx,
                         const int* __restrict__ corr1_idx,
                         const int* __restrict__ corr2_idx,
                         __bf16* __restrict__ wsA, __bf16* __restrict__ wsAl,
                         float* __restrict__ wsBias) {
  __shared__ float xs[S_];
  __shared__ float red[256];
  const int b = blockIdx.x, tid = threadIdx.x;
  const int t = b >> 4, m = b & 15;
  const float vb0 = visible_bias[0], vb1 = visible_bias[1];
  float part = 0.f;

  for (int k = tid; k < S_; k += 256) {
    float v = x[(size_t)b * S_ + k];
    xs[k] = v;
    part += v * ((k & 1) ? vb1 : vb0);
    int kt = k >> 5, ko = k & 31;
    wsA[((size_t)t * KT + kt) * FRAG_ELEMS + frag_lane(m, ko) * 16 + frag_pos(ko)] = (__bf16)v;
  }
  __syncthreads();

  const float cb0 = c0b[0], cb1 = c1b[0], cb2 = c2b[0];
  for (int c = tid; c < NP_; c += 256) {
    const int* q = corr0_idx + c * 4;
    float cv = xs[q[0]] * xs[q[1]] * xs[q[2]] * xs[q[3]];
    int K = S_ + c, kt = K >> 5, ko = K & 31;
    wsA[((size_t)t * KT + kt) * FRAG_ELEMS + frag_lane(m, ko) * 16 + frag_pos(ko)] = (__bf16)cv;
    part += cb0 * cv;
  }
  if (tid < 2 * L_) {
    int which = tid / L_, c = tid % L_;
    const int* q = (which ? corr2_idx : corr1_idx) + c * L_;
    float cv = 1.f;
    #pragma unroll
    for (int j = 0; j < L_; ++j) cv *= xs[q[j]];
    int K = S_ + NP_ + which * L_ + c, kt = K >> 5, ko = K & 31;
    wsA[((size_t)t * KT + kt) * FRAG_ELEMS + frag_lane(m, ko) * 16 + frag_pos(ko)] = (__bf16)cv;
    wsAl[((size_t)t * 2 + which) * FRAG_ELEMS + frag_lane(m, c) * 16 + frag_pos(c)] = (__bf16)cv;
    part += (which ? cb2 : cb1) * cv;
  }
  if (tid < 16) {  // K padding 1776..1791
    int K = 1776 + tid, kt = K >> 5, ko = K & 31;
    wsA[((size_t)t * KT + kt) * FRAG_ELEMS + frag_lane(m, ko) * 16 + frag_pos(ko)] = (__bf16)0.f;
  }
  if (tid < 16) {  // loop-tile K padding 24..31 for both tiles
    int which = tid >> 3, ko = 24 + (tid & 7);
    wsAl[((size_t)t * 2 + which) * FRAG_ELEMS + frag_lane(m, ko) * 16 + frag_pos(ko)] = (__bf16)0.f;
  }
  red[tid] = part;
  __syncthreads();
  for (int s = 128; s > 0; s >>= 1) { if (tid < s) red[tid] += red[tid + s]; __syncthreads(); }
  if (tid == 0) wsBias[b] = red[0];
}

// ---------------- P2: gather symmetrized weights into B fragments --------------
__global__ void k_prep_w(const float* __restrict__ sk,  const int* __restrict__ symm,
                         const float* __restrict__ c0k, const int* __restrict__ c0s,
                         const float* __restrict__ c1k, const int* __restrict__ c1s,
                         const float* __restrict__ c2k, const int* __restrict__ c2s,
                         const float* __restrict__ l0k, const float* __restrict__ l1k,
                         __bf16* __restrict__ wsB, __bf16* __restrict__ wsBl) {
  const int g = blockIdx.x, tid = threadIdx.x;
  for (int i = tid; i < KAUG * F_; i += 256) {
    int k = i >> 4, f = i & 15;
    float v = 0.f;
    if (k < S_)                       v = sk[f * S_ + symm[(size_t)g * S_ + k]];
    else if (k < S_ + NP_)            v = c0k[f * NP_ + c0s[(size_t)g * NP_ + (k - S_)]];
    else if (k < S_ + NP_ + L_)       v = c1k[f * L_ + c1s[g * L_ + (k - S_ - NP_)]];
    else if (k < S_ + NP_ + 2 * L_)   v = c2k[f * L_ + c2s[g * L_ + (k - S_ - NP_ - L_)]];
    int kt = k >> 5, ko = k & 31;
    wsB[((size_t)g * KT + kt) * FRAG_ELEMS + frag_lane(f, ko) * 16 + frag_pos(ko)] = (__bf16)v;
  }
  for (int i = tid; i < 2 * 32 * 16; i += 256) {
    int which = i >> 9, r = i & 511, k = r >> 4, f = r & 15;
    float v = 0.f;
    if (k < L_) v = which ? l1k[f * L_ + c2s[g * L_ + k]] : l0k[f * L_ + c1s[g * L_ + k]];
    wsBl[((size_t)g * 2 + which) * FRAG_ELEMS + frag_lane(f, k) * 16 + frag_pos(k)] = (__bf16)v;
  }
}

// ---------------- K3: fused WMMA + log_cosh reduction ------------------------
__global__ __launch_bounds__(128) void k_main(const __bf16* __restrict__ wsA,
                                              const __bf16* __restrict__ wsAl,
                                              const __bf16* __restrict__ wsB,
                                              const __bf16* __restrict__ wsBl,
                                              const float* __restrict__ hidden_bias,
                                              const float* __restrict__ l0b,
                                              const float* __restrict__ l1b,
                                              float* __restrict__ part) {
  __shared__ __align__(32) __bf16 sA[A_TILE_ELEMS + LOOP_TILE_ELEMS];
  const int t = blockIdx.x, gc = blockIdx.y;
  const int tid = threadIdx.x, wave = tid >> 5, lane = tid & 31;

  {  // stage whole b-tile (58 KB) into LDS, 16B vectors
    const uint4* srcA = (const uint4*)(wsA + (size_t)t * A_TILE_ELEMS);
    uint4* dA = (uint4*)sA;
    for (int i = tid; i < A_TILE_ELEMS / 8; i += 128) dA[i] = srcA[i];
    const uint4* srcL = (const uint4*)(wsAl + (size_t)t * LOOP_TILE_ELEMS);
    uint4* dL = (uint4*)(sA + A_TILE_ELEMS);
    for (int i = tid; i < LOOP_TILE_ELEMS / 8; i += 128) dL[i] = srcL[i];
  }
  __syncthreads();

  const float hb  = hidden_bias[lane & 15];
  const float lb0 = l0b[lane & 15];
  const float lb1 = l1b[lane & 15];
  // loop-term A fragments are invariant across g: load once
  const v16bf aL0 = *(const v16bf*)(sA + A_TILE_ELEMS + lane * 16);
  const v16bf aL1 = *(const v16bf*)(sA + A_TILE_ELEMS + FRAG_ELEMS + lane * 16);

  float la[8];
  #pragma unroll
  for (int j = 0; j < 8; ++j) la[j] = 0.f;

  for (int gi = 0; gi < 4; ++gi) {
    const int g = gc * 16 + wave * 4 + gi;
    const __bf16* Bg = wsB + (size_t)g * A_TILE_ELEMS;
    v8f acc = {0.f, 0.f, 0.f, 0.f, 0.f, 0.f, 0.f, 0.f};
    #pragma unroll 4
    for (int kt = 0; kt < KT; ++kt) {
      v16bf a  = *(const v16bf*)(sA + kt * FRAG_ELEMS + lane * 16);
      v16bf bv = *(const v16bf*)(Bg + kt * FRAG_ELEMS + lane * 16);
      acc = __builtin_amdgcn_wmma_f32_16x16x32_bf16(false, a, false, bv, (short)0, acc,
                                                    false, false);
    }
    #pragma unroll
    for (int j = 0; j < 8; ++j) la[j] += log_coshf(acc[j] + hb);

    {
      v8f accl = {0.f, 0.f, 0.f, 0.f, 0.f, 0.f, 0.f, 0.f};
      v16bf bv = *(const v16bf*)(wsBl + ((size_t)g * 2 + 0) * FRAG_ELEMS + lane * 16);
      accl = __builtin_amdgcn_wmma_f32_16x16x32_bf16(false, aL0, false, bv, (short)0, accl,
                                                     false, false);
      #pragma unroll
      for (int j = 0; j < 8; ++j) la[j] += log_coshf(accl[j] + lb0);
    }
    {
      v8f accl = {0.f, 0.f, 0.f, 0.f, 0.f, 0.f, 0.f, 0.f};
      v16bf bv = *(const v16bf*)(wsBl + ((size_t)g * 2 + 1) * FRAG_ELEMS + lane * 16);
      accl = __builtin_amdgcn_wmma_f32_16x16x32_bf16(false, aL1, false, bv, (short)0, accl,
                                                     false, false);
      #pragma unroll
      for (int j = 0; j < 8; ++j) la[j] += log_coshf(accl[j] + lb1);
    }
  }
  // sum over the f dimension (N = 16 lanes within each half-wave)
  #pragma unroll
  for (int j = 0; j < 8; ++j) {
    la[j] += __shfl_xor(la[j], 1, 32);
    la[j] += __shfl_xor(la[j], 2, 32);
    la[j] += __shfl_xor(la[j], 4, 32);
    la[j] += __shfl_xor(la[j], 8, 32);
  }
  if ((lane & 15) == 0) {  // lanes 0 (rows 0..7) and 16 (rows 8..15)
    const int half = lane >> 4;
    float* pp = part + (((size_t)t * GC + gc) * 4 + wave) * 16 + half * 8;
    #pragma unroll
    for (int j = 0; j < 8; ++j) pp[j] = la[j];
  }
}

// ---------------- F4: deterministic final reduction --------------------------
__global__ void k_final(const float* __restrict__ part, const float* __restrict__ bias,
                        float* __restrict__ out) {
  const int b = blockIdx.x * 256 + threadIdx.x;
  if (b >= B_) return;
  const int t = b >> 4, row = b & 15;
  float s = bias[b];
  const float* p = part + (size_t)t * GC * 4 * 16 + row;
  for (int i = 0; i < GC * 4; ++i) s += p[i * 16];
  out[b] = s;
}

extern "C" void kernel_launch(void* const* d_in, const int* in_sizes, int n_in,
                              void* d_out, int out_size, void* d_ws, size_t ws_size,
                              hipStream_t stream) {
  (void)in_sizes; (void)n_in; (void)out_size; (void)ws_size;
  const float* x   = (const float*)d_in[0];
  const float* hb  = (const float*)d_in[1];
  const float* sk  = (const float*)d_in[2];
  const float* vb  = (const float*)d_in[3];
  const float* c0b = (const float*)d_in[4];
  const float* c0k = (const float*)d_in[5];
  const float* c1b = (const float*)d_in[6];
  const float* c1k = (const float*)d_in[7];
  const float* c2b = (const float*)d_in[8];
  const float* c2k = (const float*)d_in[9];
  const float* l0b = (const float*)d_in[10];
  const float* l0k = (const float*)d_in[11];
  const float* l1b = (const float*)d_in[12];
  const float* l1k = (const float*)d_in[13];
  const int* symm = (const int*)d_in[14];
  const int* c0i  = (const int*)d_in[15];
  const int* c0s  = (const int*)d_in[16];
  const int* c1i  = (const int*)d_in[17];
  const int* c1s  = (const int*)d_in[18];
  const int* c2i  = (const int*)d_in[19];
  const int* c2s  = (const int*)d_in[20];

  char* ws = (char*)d_ws;
  __bf16* wsA   = (__bf16*)(ws + WS_A);
  __bf16* wsAl  = (__bf16*)(ws + WS_AL);
  __bf16* wsB   = (__bf16*)(ws + WS_B);
  __bf16* wsBl  = (__bf16*)(ws + WS_BL);
  float*  part  = (float*)(ws + WS_PART);
  float*  wbias = (float*)(ws + WS_BIAS);
  float*  out   = (float*)d_out;

  hipLaunchKernelGGL(k_prep_x, dim3(B_), dim3(256), 0, stream,
                     x, vb, c0b, c1b, c2b, c0i, c1i, c2i, wsA, wsAl, wbias);
  hipLaunchKernelGGL(k_prep_w, dim3(G_), dim3(256), 0, stream,
                     sk, symm, c0k, c0s, c1k, c1s, c2k, c2s, l0k, l1k, wsB, wsBl);
  hipLaunchKernelGGL(k_main, dim3(TILEB, GC), dim3(128), 0, stream,
                     wsA, wsAl, wsB, wsBl, hb, l0b, l1b, part);
  hipLaunchKernelGGL(k_final, dim3(B_ / 256), dim3(256), 0, stream,
                     part, wbias, out);
}
